// SeqCondAttention_61529701483094
// MI455X (gfx1250) — matrix operand order repaired and checked by reading, not compiled, not run on hardware
//
#include <hip/hip_runtime.h>
#include <hip/hip_bf16.h>

// ---------------------------------------------------------------------------
// Problem constants (match reference)
// ---------------------------------------------------------------------------
#define BB      2
#define LL      2048
#define DD      1024
#define KK      32
#define MM      4
#define AA      4
#define HH      32            // DD / KK
#define CKK     4
#define TOTAL   2080          // 2*DD + KK
#define HM      128           // HH * MM
#define NSEG    16
#define SEGL    128           // LL / NSEG
#define PSTRIDE 260           // 128 re + 128 im + 1 den, padded
#define NT      4             // GEMM: 16-col tiles per wave (64 N per wave)

typedef __attribute__((ext_vector_type(16))) __bf16 v16bf;
typedef __attribute__((ext_vector_type(8)))  float  v8f;

union BfVec { unsigned u[8]; v16bf v; };

__device__ __forceinline__ unsigned short f2bf(float f) {
  union { float f; unsigned u; } cv; cv.f = f;
  unsigned u = cv.u;
  unsigned r = (u + 0x7FFFu + ((u >> 16) & 1u)) >> 16;  // round-nearest-even
  return (unsigned short)r;
}

__device__ __forceinline__ float softplusf(float x) {
  return logf(1.0f + __expf(x));
}

// WMMA operand fragment gathers (bf16, 16x16x32 shapes).
// A 16x32 layout (ISA table):  lane row = lane&15;  k = (v>>2)*16 + hi*8 + (v&3)*2
// B 32x16 layout (ISA table):  lane col = lane&15;  k = (v>>2)*16 + hi*8 + (v&3)*2
__device__ __forceinline__ v16bf frag_a(const unsigned short* row, int kc, int hi) {
  BfVec a;
#pragma unroll
  for (int v = 0; v < 8; ++v) {
    const int k = kc + ((v >> 2) << 4) + (hi << 3) + ((v & 3) << 1);
    a.u[v] = *(const unsigned*)(row + k);
  }
  return a.v;
}
__device__ __forceinline__ v16bf frag_b(const unsigned short* col, int kc, int hi) {
  BfVec b;
#pragma unroll
  for (int v = 0; v < 8; ++v) {
    const int k = kc + ((v >> 2) << 4) + (hi << 3) + ((v & 3) << 1);
    b.u[v] = *(const unsigned*)(col + k);
  }
  return b.v;
}

// ---------------------------------------------------------------------------
// Cast helpers
// ---------------------------------------------------------------------------
__global__ void cast_f32_bf16_kernel(const float* __restrict__ src,
                                     unsigned short* __restrict__ dst, int n) {
  int i = blockIdx.x * blockDim.x + threadIdx.x;
  if (i < n) dst[i] = f2bf(src[i]);
}

// src[rows][cols] (f32, row-major) -> dst[cols][rows] (bf16, row-major)
__global__ void cast_transpose_bf16_kernel(const float* __restrict__ src,
                                           unsigned short* __restrict__ dst,
                                           int rows, int cols) {
  int i = blockIdx.x * blockDim.x + threadIdx.x;
  if (i >= rows * cols) return;
  int r = i / cols, c = i % cols;
  dst[(size_t)c * rows + r] = f2bf(src[i]);
}

// ---------------------------------------------------------------------------
// bf16 WMMA GEMM:  C[M x N] = A[M x K] * B[K x N]
//   A  : bf16 row-major (lda = K);  BT : B transposed, BT[n][k] (ldb = K)
//   C  : f32 row-major (ldc = N)
// One wave computes a 16 x (16*NT) strip: one A fragment feeds NT WMMAs per
// 32-wide K chunk (4x A reuse). M,K multiples of 16/32; N may be ragged.
// ---------------------------------------------------------------------------
__global__ __launch_bounds__(32)
void gemm_bf16_wmma_kernel(const unsigned short* __restrict__ A,
                           const unsigned short* __restrict__ BT,
                           float* __restrict__ C,
                           int Mdim, int Ndim, int Kdim) {
  const int lane = threadIdx.x & 31;
  const int l15  = lane & 15;
  const int hi   = lane >> 4;
  const int m0   = blockIdx.y * 16;
  const int n0   = blockIdx.x * (16 * NT);

  const unsigned short* arow = A + (size_t)(m0 + l15) * Kdim;
  const unsigned short* brow[NT];
  bool nok[NT];
#pragma unroll
  for (int t = 0; t < NT; ++t) {
    int nc = n0 + t * 16 + l15;
    nok[t] = (nc < Ndim);
    if (!nok[t]) nc = Ndim - 1;   // clamp: keep EXEC all-ones at the WMMA
    brow[t] = BT + (size_t)nc * Kdim;
  }

  v8f acc[NT] = {};
  for (int k0 = 0; k0 < Kdim; k0 += 32) {
    if (k0 + 32 < Kdim) {                       // -> global_prefetch_b8
      __builtin_prefetch(arow + k0 + 32, 0, 3);
#pragma unroll
      for (int t = 0; t < NT; ++t) __builtin_prefetch(brow[t] + k0 + 32, 0, 3);
    }
    const v16bf a = frag_a(arow, k0, hi);
#pragma unroll
    for (int t = 0; t < NT; ++t) {
      const v16bf b = frag_b(brow[t], k0, hi);
      acc[t] = __builtin_amdgcn_wmma_f32_16x16x32_bf16(
          false, a, false, b, (short)0, acc[t], false, false);
    }
  }

#pragma unroll
  for (int t = 0; t < NT; ++t) {
    if (!nok[t]) continue;
#pragma unroll
    for (int r = 0; r < 8; ++r) {
      // C/D layout: lane&15 = N col, VGPR r = row r (lanes 0-15) / 8+r (16-31)
      C[(size_t)(m0 + r + 8 * hi) * Ndim + (n0 + t * 16 + l15)] = acc[t][r];
    }
  }
}

// ---------------------------------------------------------------------------
// Depthwise causal conv (CK taps) + bias:  z = conv(z_pre) + b
// ---------------------------------------------------------------------------
__global__ void conv_bias_kernel(const float* __restrict__ zp,
                                 const float* __restrict__ cw,
                                 const float* __restrict__ cb,
                                 float* __restrict__ z, int n) {
  int idx = blockIdx.x * blockDim.x + threadIdx.x;
  if (idx >= n) return;
  int c  = idx % TOTAL;
  int bl = idx / TOTAL;
  int l  = bl % LL;
  int b  = bl / LL;
  float acc = cb[c];
#pragma unroll
  for (int t = 0; t < CKK; ++t) {
    int ls = l - (CKK - 1) + t;
    if (ls >= 0)
      acc += cw[t * TOTAL + c] * zp[((size_t)b * LL + ls) * TOTAL + c];
  }
  z[idx] = acc;
}

// ---------------------------------------------------------------------------
// Scan step (shared by pass A and pass C). One wave handles one (b,k); lane =
// head-dim index h. Each lane owns features f = lane*4 + m (m = 0..3).
// den is a scalar per (b,l,k) (broadcast block of `merged`) -> replicated.
// ---------------------------------------------------------------------------
__device__ __forceinline__ void scan_step(
    const float* __restrict__ z, size_t zrow, int k, int lane, int l,
    float sscale, float slope, bool anchor, const float th[4],
    float w0, float w1, float w2,
    float accRe[4], float accIm[4], float& den) {
  float xv = z[zrow + k * HH + lane];
  float s  = z[zrow + 2 * DD + k];
  float sc = fminf(fmaxf(sscale * s, -20.f), 20.f);
  float p  = __expf(sc);
  float tw = anchor ? __expf(-slope * (float)l)
                    : __expf(-slope * (float)(LL - 1 - l));
  float pw = p * tw;
  den += pw;
  float poly = w0 + w1 * xv - w2 * xv * xv;
  float ppw  = pw * poly;
#pragma unroll
  for (int m = 0; m < MM; ++m) {
    float sn, cs;
    __sincosf(xv * th[m], &sn, &cs);
    accRe[m] += ppw * cs;
    accIm[m] += ppw * sn;
  }
}

__device__ __forceinline__ void scan_setup(
    const float* __restrict__ theta, const float* __restrict__ decay_slopes,
    const float* __restrict__ anchor_slopes, const float* __restrict__ score_scale,
    const float* __restrict__ deriv_logits, int k,
    float th[4], float& sscale, float& slope, bool& anchor,
    float& w0, float& w1, float& w2) {
#pragma unroll
  for (int m = 0; m < MM; ++m) th[m] = theta[m];  // theta identical across k,h
  float e0 = __expf(deriv_logits[0]);
  float e1 = __expf(deriv_logits[1]);
  float e2 = __expf(deriv_logits[2]);
  float inv = 1.0f / (e0 + e1 + e2);
  w0 = e0 * inv; w1 = e1 * inv; w2 = e2 * inv;
  anchor = (k >= KK - AA);
  slope  = anchor ? softplusf(anchor_slopes[k - (KK - AA)])
                  : softplusf(decay_slopes[k]);
  sscale = score_scale[k];
}

// Pass A: per (b,k,seg) segment totals -> partials[(b*K+k)*NSEG+seg][0..256]
__global__ __launch_bounds__(32)
void scan_partials_kernel(const float* __restrict__ z,
                          const float* __restrict__ theta,
                          const float* __restrict__ decay_slopes,
                          const float* __restrict__ anchor_slopes,
                          const float* __restrict__ score_scale,
                          const float* __restrict__ deriv_logits,
                          float* __restrict__ partials) {
  int blk = blockIdx.x;
  int seg = blk % NSEG;
  int k   = (blk / NSEG) % KK;
  int b   = blk / (NSEG * KK);
  int lane = threadIdx.x;

  float th[4], sscale, slope, w0, w1, w2; bool anchor;
  scan_setup(theta, decay_slopes, anchor_slopes, score_scale, deriv_logits, k,
             th, sscale, slope, anchor, w0, w1, w2);

  float accRe[4] = {0.f, 0.f, 0.f, 0.f};
  float accIm[4] = {0.f, 0.f, 0.f, 0.f};
  float den = 0.f;

  size_t zrow = ((size_t)b * LL + seg * SEGL) * TOTAL;
  for (int l = seg * SEGL; l < (seg + 1) * SEGL; ++l, zrow += TOTAL)
    scan_step(z, zrow, k, lane, l, sscale, slope, anchor, th, w0, w1, w2,
              accRe, accIm, den);

  float* P = partials + ((size_t)(b * KK + k) * NSEG + seg) * PSTRIDE;
#pragma unroll
  for (int m = 0; m < MM; ++m) {
    P[lane * 4 + m]      = accRe[m];
    P[HM + lane * 4 + m] = accIm[m];
  }
  if (lane == 0) P[2 * HM] = den;
}

// Pass B: in-place exclusive prefix over the NSEG axis, per (b,k)
__global__ __launch_bounds__(32)
void seg_prefix_kernel(float* __restrict__ partials) {
  int bk   = blockIdx.x;
  int lane = threadIdx.x;
  for (int j = lane; j < 2 * HM + 1; j += 32) {
    float* base = partials + (size_t)bk * NSEG * PSTRIDE + j;
    float run = 0.f;
    for (int s = 0; s < NSEG; ++s) {
      float t = base[(size_t)s * PSTRIDE];
      base[(size_t)s * PSTRIDE] = run;
      run += t;
    }
  }
}

// ---------------------------------------------------------------------------
// Pass C: replay each 128-step segment from its exclusive prefix.
// The sequential part (scan accumulate + RMS norm) runs per step; the
// (16 x 128) @ (128 x 32) W_re/W_im matmul is batched over 16 timesteps and
// executed on the WMMA pipes from LDS-staged bf16 fragments. SiLU gate and
// bf16 emission are fused into the tile unpack.
// ---------------------------------------------------------------------------
__global__ __launch_bounds__(32)
void scan_emit_kernel(const float* __restrict__ z,
                      const float* __restrict__ theta,
                      const float* __restrict__ decay_slopes,
                      const float* __restrict__ anchor_slopes,
                      const float* __restrict__ score_scale,
                      const float* __restrict__ deriv_logits,
                      const float* __restrict__ norm_scale,
                      const float* __restrict__ W_re,
                      const float* __restrict__ W_im,
                      const float* __restrict__ partials,
                      unsigned short* __restrict__ yg) {
  __shared__ unsigned short sWreT[HH * HM];   // W_re^T [h'][f], bf16
  __shared__ unsigned short sWimT[HH * HM];   // W_im^T [h'][f], bf16
  __shared__ unsigned short sRe16[16 * HM];   // 16 steps of rn_re, bf16
  __shared__ unsigned short sIm16[16 * HM];   // 16 steps of rn_im, bf16

  int blk = blockIdx.x;
  int seg = blk % NSEG;
  int k   = (blk / NSEG) % KK;
  int b   = blk / (NSEG * KK);
  int lane = threadIdx.x;
  const int l15 = lane & 15;
  const int hi  = lane >> 4;

  // preload transposed bf16 W into LDS
  for (int i = lane; i < HH * HM; i += 32) {
    int h = i / HM, f = i % HM;
    sWreT[i] = f2bf(W_re[f * HH + h]);
    sWimT[i] = f2bf(W_im[f * HH + h]);
  }
  __syncthreads();

  float th[4], sscale, slope, w0, w1, w2; bool anchor;
  scan_setup(theta, decay_slopes, anchor_slopes, score_scale, deriv_logits, k,
             th, sscale, slope, anchor, w0, w1, w2);

  const float* P = partials + ((size_t)(b * KK + k) * NSEG + seg) * PSTRIDE;
  float accRe[4], accIm[4], nsRe[4], nsIm[4];
#pragma unroll
  for (int m = 0; m < MM; ++m) {
    accRe[m] = P[lane * 4 + m];
    accIm[m] = P[HM + lane * 4 + m];
    nsRe[m]  = norm_scale[lane * 4 + m];
    nsIm[m]  = norm_scale[HM + lane * 4 + m];
  }
  float den = P[2 * HM];

  const size_t zbase = (size_t)b * LL * TOTAL;
  for (int l0 = seg * SEGL; l0 < (seg + 1) * SEGL; l0 += 16) {
    // --- sequential phase: 16 scan steps + per-step RMS norm -> LDS bf16 ---
    for (int sl = 0; sl < 16; ++sl) {
      const int l = l0 + sl;
      scan_step(z, zbase + (size_t)l * TOTAL, k, lane, l,
                sscale, slope, anchor, th, w0, w1, w2, accRe, accIm, den);

      float inv = 1.0f / fmaxf(den, 1e-4f);
      float part = 0.f;
      float rr[4], ii[4];
#pragma unroll
      for (int m = 0; m < MM; ++m) {
        rr[m] = accRe[m] * inv;
        ii[m] = accIm[m] * inv;
        part += rr[m] * rr[m] + ii[m] * ii[m];
      }
#pragma unroll
      for (int off = 16; off > 0; off >>= 1)
        part += __shfl_xor(part, off, 32);
      float rs = rsqrtf(part * (1.0f / (2.0f * HM)) + 1e-5f);

      unsigned short* pr = &sRe16[sl * HM + lane * 4];
      unsigned short* pi = &sIm16[sl * HM + lane * 4];
#pragma unroll
      for (int m = 0; m < MM; ++m) {
        pr[m] = f2bf(rr[m] * rs * nsRe[m]);
        pi[m] = f2bf(ii[m] * rs * nsIm[m]);
      }
    }
    __syncthreads();

    // --- matrix phase: y[16 x 32] = rn_re @ W_re + rn_im @ W_im (WMMA) ---
    v8f accT[2] = {};
#pragma unroll
    for (int kc = 0; kc < HM; kc += 32) {
      const v16bf aRe = frag_a(&sRe16[l15 * HM], kc, hi);
      const v16bf aIm = frag_a(&sIm16[l15 * HM], kc, hi);
#pragma unroll
      for (int t = 0; t < 2; ++t) {
        const v16bf bRe = frag_b(&sWreT[(t * 16 + l15) * HM], kc, hi);
        accT[t] = __builtin_amdgcn_wmma_f32_16x16x32_bf16(
            false, aRe, false, bRe, (short)0, accT[t], false, false);
        const v16bf bIm = frag_b(&sWimT[(t * 16 + l15) * HM], kc, hi);
        accT[t] = __builtin_amdgcn_wmma_f32_16x16x32_bf16(
            false, aIm, false, bIm, (short)0, accT[t], false, false);
      }
    }

    // --- unpack: SiLU gate + bf16 store for the out-proj GEMM ---
#pragma unroll
    for (int t = 0; t < 2; ++t) {
#pragma unroll
      for (int r = 0; r < 8; ++r) {
        const int step = r + 8 * hi;          // C/D row layout
        const int l    = l0 + step;
        const int col  = t * 16 + l15;        // h' index
        float g = z[zbase + (size_t)l * TOTAL + DD + k * HH + col];
        float gate = g / (1.0f + __expf(-g));
        yg[((size_t)b * LL + l) * DD + k * HH + col] = f2bf(accT[t][r] * gate);
      }
    }
    __syncthreads();   // rn buffers reused next 16-step block
  }
}

// ---------------------------------------------------------------------------
// Launch
// ---------------------------------------------------------------------------
extern "C" void kernel_launch(void* const* d_in, const int* in_sizes, int n_in,
                              void* d_out, int out_size, void* d_ws, size_t ws_size,
                              hipStream_t stream) {
  (void)in_sizes; (void)n_in; (void)out_size; (void)ws_size;

  const float* x             = (const float*)d_in[0];
  const float* in_proj_w     = (const float*)d_in[1];
  const float* conv_w        = (const float*)d_in[2];
  const float* conv_b        = (const float*)d_in[3];
  const float* theta         = (const float*)d_in[4];
  const float* decay_slopes  = (const float*)d_in[5];
  const float* anchor_slopes = (const float*)d_in[6];
  const float* score_scale   = (const float*)d_in[7];
  const float* deriv_logits  = (const float*)d_in[8];
  const float* norm_scale    = (const float*)d_in[9];
  const float* W_re          = (const float*)d_in[10];
  const float* W_im          = (const float*)d_in[11];
  const float* out_proj_w    = (const float*)d_in[12];
  float*       out           = (float*)d_out;

  const int BL = BB * LL;  // 4096 rows

  // Workspace carve-up (~93 MB total)
  char* ws = (char*)d_ws;
  size_t off = 0;
  auto carve = [&](size_t bytes) {
    size_t o = off;
    off = (off + bytes + 255) & ~(size_t)255;
    return o;
  };
  unsigned short* xb16  = (unsigned short*)(ws + carve((size_t)BL * DD * 2));
  unsigned short* wT16  = (unsigned short*)(ws + carve((size_t)TOTAL * DD * 2));
  unsigned short* opT16 = (unsigned short*)(ws + carve((size_t)DD * DD * 2));
  float*          zpre  = (float*)(ws + carve((size_t)BL * TOTAL * 4));
  float*          zbuf  = (float*)(ws + carve((size_t)BL * TOTAL * 4));
  unsigned short* yg16  = (unsigned short*)(ws + carve((size_t)BL * DD * 2));
  float*          parts = (float*)(ws + carve((size_t)BB * KK * NSEG * PSTRIDE * 4));

  // 1. casts / transposes to bf16
  {
    int n = BL * DD;
    cast_f32_bf16_kernel<<<(n + 255) / 256, 256, 0, stream>>>(x, xb16, n);
  }
  {
    int n = DD * TOTAL;
    cast_transpose_bf16_kernel<<<(n + 255) / 256, 256, 0, stream>>>(
        in_proj_w, wT16, DD, TOTAL);
  }
  {
    int n = DD * DD;
    cast_transpose_bf16_kernel<<<(n + 255) / 256, 256, 0, stream>>>(
        out_proj_w, opT16, DD, DD);
  }

  // 2. in-proj GEMM (WMMA):  zpre[BL x TOTAL] = xb16 @ in_proj_w
  {
    dim3 grid((TOTAL + 16 * NT - 1) / (16 * NT), BL / 16);
    gemm_bf16_wmma_kernel<<<grid, 32, 0, stream>>>(xb16, wT16, zpre,
                                                   BL, TOTAL, DD);
  }

  // 3. depthwise causal conv + bias
  {
    int n = BL * TOTAL;
    conv_bias_kernel<<<(n + 255) / 256, 256, 0, stream>>>(zpre, conv_w, conv_b,
                                                          zbuf, n);
  }

  // 4. segment-parallel cumulative scan (3 passes) + norm + head matmul + gate
  scan_partials_kernel<<<BB * KK * NSEG, 32, 0, stream>>>(
      zbuf, theta, decay_slopes, anchor_slopes, score_scale, deriv_logits,
      parts);
  seg_prefix_kernel<<<BB * KK, 32, 0, stream>>>(parts);
  scan_emit_kernel<<<BB * KK * NSEG, 32, 0, stream>>>(
      zbuf, theta, decay_slopes, anchor_slopes, score_scale, deriv_logits,
      norm_scale, W_re, W_im, parts, yg16);

  // 5. out-proj GEMM (WMMA):  out[BL x DD] = yg16 @ out_proj_w
  {
    dim3 grid(DD / (16 * NT), BL / 16);
    gemm_bf16_wmma_kernel<<<grid, 32, 0, stream>>>(yg16, opT16, out,
                                                   BL, DD, DD);
  }
}
